// SinCosGraphRNN_53652731462309
// MI455X (gfx1250) — matrix operand order repaired and checked by reading, not compiled, not run on hardware
//
#include <hip/hip_runtime.h>
#include <hip/hip_bf16.h>
#include <math.h>

typedef _Float16 half_t;
typedef __attribute__((ext_vector_type(16))) _Float16 v16h;
typedef __attribute__((ext_vector_type(8)))  _Float16 v8h;
typedef __attribute__((ext_vector_type(4)))  _Float16 v4h;
typedef __attribute__((ext_vector_type(8)))  float    v8f;
typedef __attribute__((ext_vector_type(4)))  float    v4f;

#define T_ 8
#define N_ 4096
#define F_ 64
#define PD_ 64
#define H_ 128
#define TN_ (T_ * N_)
#define EC_ 25600   // edge chunk rows (multiple of 128)

// ---------------------------------------------------------------------------
// Small helper kernels
// ---------------------------------------------------------------------------

__global__ void cvt_f16_kernel(const float* __restrict__ src, half_t* __restrict__ dst, int n) {
    int i = blockIdx.x * 256 + threadIdx.x;
    if (i < n) dst[i] = (half_t)src[i];
}

__global__ void pos_kernel(half_t* __restrict__ pos_h) {
    int idx = blockIdx.x * 256 + threadIdx.x;
    if (idx >= N_ * PD_) return;
    int n = idx / PD_, c = idx % PD_;
    float div = __expf(-(float)(c & ~1) * (logf(10000.0f) / (float)PD_));
    float ang = (float)n * div;
    float v = (c & 1) ? cosf(ang) : sinf(ang);
    pos_h[idx] = (half_t)v;
}

// x = concat(node_features(f32, 64), pos(64)) -> f16 (TN x 128)
__global__ void concat_enc_kernel(const float* __restrict__ nf, const half_t* __restrict__ pos_h,
                                  half_t* __restrict__ x) {
    int idx = blockIdx.x * 256 + threadIdx.x;
    if (idx >= TN_ * 128) return;
    int r = idx >> 7, c = idx & 127;
    x[idx] = (c < 64) ? (half_t)nf[(size_t)r * 64 + c]
                      : pos_h[(size_t)(r % N_) * 64 + (c - 64)];
}

// mi = concat(h(128), pos(64)) -> (TN x 192)
__global__ void concat_hpos_kernel(const half_t* __restrict__ hh, const half_t* __restrict__ pos_h,
                                   half_t* __restrict__ mi) {
    long idx = (long)blockIdx.x * 256 + threadIdx.x;
    if (idx >= (long)TN_ * 192) return;
    int r = (int)(idx / 192), c = (int)(idx % 192);
    mi[idx] = (c < 128) ? hh[(size_t)r * 128 + c]
                        : pos_h[(size_t)(r % N_) * 64 + (c - 128)];
}

// ui = concat(h(128), agg(128)) -> (TN x 256)
__global__ void concat_hagg_kernel(const half_t* __restrict__ hh, const half_t* __restrict__ aggh,
                                   half_t* __restrict__ ui) {
    long idx = (long)blockIdx.x * 256 + threadIdx.x;
    if (idx >= (long)TN_ * 256) return;
    int r = (int)(idx >> 8), c = (int)(idx & 255);
    ui[idx] = (c < 128) ? hh[(size_t)r * 128 + c]
                        : aggh[(size_t)r * 128 + (c - 128)];
}

// ---------------------------------------------------------------------------
// Generic tiled WMMA GEMM: C(MxN) = relu?(A(MxK) * B(KxN) + bias)
// NT = N/16 compile-time (register-resident accumulators).
// Double-buffered LDS B tile (software pipeline) + register-pipelined
// B-fragment loads. A: f16 row-major (lda); B: f16 KxN row-major; K%32==0.
// Block = 256 threads = 8 wave32; block computes 128 rows x N cols.
// ---------------------------------------------------------------------------
template <int NT>
__global__ void gemm_f16_kernel(const half_t* __restrict__ A, int lda,
                                const half_t* __restrict__ B,
                                const float* __restrict__ bias,
                                float* __restrict__ C,
                                half_t* __restrict__ Ch,
                                int M, int K, int relu) {
    constexpr int N = NT * 16;
    __shared__ half_t sBT[2][128 * 32];        // [n][k] transposed B tiles
    const int lane = threadIdx.x & 31;
    const int wid  = threadIdx.x >> 5;
    const int g    = lane >> 4;
    const int mrow = lane & 15;
    const int m0   = blockIdx.x * 128 + wid * 16;

    // staging map: this thread owns k quad [kq,kq+4) x n quad [n0,n0+4)
    const int kq = (threadIdx.x & 7) * 4;
    const int n0 = (threadIdx.x >> 3) * 4;

    v4h st[4];
    auto stage = [&](int k0) {
        if (n0 < N) {
            const half_t* src = B + (size_t)(k0 + kq) * N + n0;
#pragma unroll
            for (int kk = 0; kk < 4; ++kk) st[kk] = *(const v4h*)(src + kk * N);
        }
    };
    auto commit = [&](int buf) {
        if (n0 < N) {
#pragma unroll
            for (int i = 0; i < 4; ++i) {
                v4h col;
#pragma unroll
                for (int kk = 0; kk < 4; ++kk) col[kk] = st[kk][i];
                *(v4h*)&sBT[buf][(n0 + i) * 32 + kq] = col;   // ds_store_b64
            }
        }
    };

    v8f acc[NT];
#pragma unroll
    for (int nt = 0; nt < NT; ++nt) acc[nt] = (v8f)0.0f;

    stage(0);
    commit(0);
    __syncthreads();

    for (int k0 = 0; k0 < K; k0 += 32) {
        const int buf = (k0 >> 5) & 1;
        const bool more = (k0 + 32) < K;
        if (more) stage(k0 + 32);              // global loads, overlapped

        // A fragment: 16x32, M = lane%16; K halves split by lane group
        const half_t* arow = A + (size_t)(m0 + mrow) * lda + k0;
        v16h a;
#pragma unroll
        for (int i = 0; i < 8; ++i) a[i] = arow[16 * g + i];
#pragma unroll
        for (int i = 0; i < 8; ++i) a[8 + i] = arow[16 * g + 8 + i];

        if (more) commit(buf ^ 1);             // write other buffer (safe)

        const half_t* bbase = &sBT[buf][mrow * 32 + 16 * g];
        v16h bcur;
#pragma unroll
        for (int i = 0; i < 16; ++i) bcur[i] = bbase[i];
#pragma unroll
        for (int nt = 0; nt < NT; ++nt) {
            v16h bnext;
            if (nt + 1 < NT) {
                const half_t* bp = bbase + (size_t)(nt + 1) * 16 * 32;
#pragma unroll
                for (int i = 0; i < 16; ++i) bnext[i] = bp[i];
            }
            acc[nt] = __builtin_amdgcn_wmma_f32_16x16x32_f16(
                false, a, false, bcur, (short)0, acc[nt], false, false);
            bcur = bnext;
        }
        __syncthreads();
    }
    // epilogue: C layout VGPR r -> M = r + 8*g, N = lane%16
#pragma unroll
    for (int nt = 0; nt < NT; ++nt) {
        int nn = nt * 16 + (lane & 15);
        float bv = bias ? bias[nn] : 0.0f;
#pragma unroll
        for (int r = 0; r < 8; ++r) {
            int mm = m0 + 8 * g + r;
            if (mm < M) {
                float v = acc[nt][r] + bv;
                if (relu) v = v > 0.0f ? v : 0.0f;
                if (C)  C[(size_t)mm * N + nn] = v;
                if (Ch) Ch[(size_t)mm * N + nn] = (half_t)v;
            }
        }
    }
}

// ---------------------------------------------------------------------------
// Aggregation: agg[t] = adj[t](NxN f32, 0/1) @ m[t](NxH f16) -> f16
// Streams adj (f32 -> packed f16 cvt) and m through double-buffered LDS,
// WMMA f16 with f32 accumulate. Grid: (N/128, T), 256 threads.
// ---------------------------------------------------------------------------
__global__ void agg_kernel(const float* __restrict__ adj,
                           const half_t* __restrict__ mh,
                           half_t* __restrict__ aggh) {
    __shared__ half_t sA[2][128 * 32];
    __shared__ half_t sBT[2][128 * 32];
    const int t    = blockIdx.y;
    const int lane = threadIdx.x & 31;
    const int wid  = threadIdx.x >> 5;
    const int g    = lane >> 4;
    const int mrow = lane & 15;
    const int row0 = blockIdx.x * 128;
    const float*  adjT = adj + (size_t)t * N_ * N_;
    const half_t* mT   = mh  + (size_t)t * N_ * H_;

    // adj staging: thread owns row r, 16 consecutive k at kc
    const int r  = threadIdx.x >> 1;
    const int kc = (threadIdx.x & 1) * 16;
    // m staging: k quad x n quad as in gemm (N = 128)
    const int kq = (threadIdx.x & 7) * 4;
    const int n0 = (threadIdx.x >> 3) * 4;

    v4f fa[4];
    v4h sb[4];
    auto stage = [&](int k0) {
        const float* asrc = adjT + (size_t)(row0 + r) * N_ + k0 + kc;
#pragma unroll
        for (int j = 0; j < 4; ++j) fa[j] = *(const v4f*)(asrc + 4 * j);
        if (k0 + 32 < N_)
            __builtin_prefetch(asrc + 32, 0, 0);
        const half_t* bsrc = mT + (size_t)(k0 + kq) * H_ + n0;
#pragma unroll
        for (int kk = 0; kk < 4; ++kk) sb[kk] = *(const v4h*)(bsrc + kk * H_);
    };
    auto commit = [&](int buf) {
        v8h lo, hi;
#pragma unroll
        for (int i = 0; i < 8; ++i) {
            lo[i] = (half_t)fa[i >> 2][i & 3];
            hi[i] = (half_t)fa[2 + (i >> 2)][i & 3];
        }
        *(v8h*)&sA[buf][r * 32 + kc]     = lo;      // ds_store_b128
        *(v8h*)&sA[buf][r * 32 + kc + 8] = hi;      // ds_store_b128
#pragma unroll
        for (int i = 0; i < 4; ++i) {
            v4h col;
#pragma unroll
            for (int kk = 0; kk < 4; ++kk) col[kk] = sb[kk][i];
            *(v4h*)&sBT[buf][(n0 + i) * 32 + kq] = col;  // ds_store_b64
        }
    };

    v8f acc[8];
#pragma unroll
    for (int nt = 0; nt < 8; ++nt) acc[nt] = (v8f)0.0f;

    stage(0);
    commit(0);
    __syncthreads();

    for (int k0 = 0; k0 < N_; k0 += 32) {
        const int buf = (k0 >> 5) & 1;
        const bool more = (k0 + 32) < N_;
        if (more) stage(k0 + 32);

        if (more) commit(buf ^ 1);

        const half_t* arow = &sA[buf][(wid * 16 + mrow) * 32];
        v16h a;
#pragma unroll
        for (int i = 0; i < 8; ++i) a[i] = arow[16 * g + i];
#pragma unroll
        for (int i = 0; i < 8; ++i) a[8 + i] = arow[16 * g + 8 + i];

        const half_t* bbase = &sBT[buf][mrow * 32 + 16 * g];
        v16h bcur;
#pragma unroll
        for (int i = 0; i < 16; ++i) bcur[i] = bbase[i];
#pragma unroll
        for (int nt = 0; nt < 8; ++nt) {
            v16h bnext;
            if (nt + 1 < 8) {
                const half_t* bp = bbase + (size_t)(nt + 1) * 16 * 32;
#pragma unroll
                for (int i = 0; i < 16; ++i) bnext[i] = bp[i];
            }
            acc[nt] = __builtin_amdgcn_wmma_f32_16x16x32_f16(
                false, a, false, bcur, (short)0, acc[nt], false, false);
            bcur = bnext;
        }
        __syncthreads();
    }
#pragma unroll
    for (int nt = 0; nt < 8; ++nt) {
        int nn = nt * 16 + (lane & 15);
#pragma unroll
        for (int rr = 0; rr < 8; ++rr) {
            int mm = row0 + wid * 16 + 8 * g + rr;
            aggh[((size_t)t * N_ + mm) * H_ + nn] = (half_t)acc[nt][rr];
        }
    }
}

// ---------------------------------------------------------------------------
// h = LayerNorm(u + h) * gamma + beta ; one wave32 per row (128 cols)
// ---------------------------------------------------------------------------
__global__ void ln_residual_kernel(const float* __restrict__ u,
                                   float* __restrict__ h,
                                   half_t* __restrict__ hh,
                                   const float* __restrict__ gamma,
                                   const float* __restrict__ beta) {
    int lane = threadIdx.x & 31;
    int wid  = threadIdx.x >> 5;
    int row  = blockIdx.x * 8 + wid;
    const float* ur = u + (size_t)row * 128;
    float* hr = h + (size_t)row * 128;

    float x[4], s = 0.0f;
#pragma unroll
    for (int i = 0; i < 4; ++i) {
        x[i] = ur[lane * 4 + i] + hr[lane * 4 + i];
        s += x[i];
    }
#pragma unroll
    for (int m = 16; m >= 1; m >>= 1) s += __shfl_xor(s, m, 32);
    float mu = s * (1.0f / 128.0f);
    float v = 0.0f;
#pragma unroll
    for (int i = 0; i < 4; ++i) { float d = x[i] - mu; v += d * d; }
#pragma unroll
    for (int m = 16; m >= 1; m >>= 1) v += __shfl_xor(v, m, 32);
    float inv = rsqrtf(v * (1.0f / 128.0f) + 1e-5f);
#pragma unroll
    for (int i = 0; i < 4; ++i) {
        int c = lane * 4 + i;
        float y = (x[i] - mu) * inv * gamma[c] + beta[c];
        hr[c] = y;
        hh[(size_t)row * 128 + c] = (half_t)y;
    }
}

__global__ void pool_kernel(const float* __restrict__ h, float* __restrict__ pooled) {
    int t = blockIdx.x, d = threadIdx.x;          // 128 threads
    const float* base = h + (size_t)t * N_ * 128 + d;
    float s = 0.0f;
    for (int n = 0; n < N_; ++n) s += base[(size_t)n * 128];
    pooled[t * 128 + d] = s * (1.0f / (float)N_);
}

__device__ __forceinline__ float sigmf(float x) { return 1.0f / (1.0f + __expf(-x)); }

// pooled MLP + 2-layer LSTM over T=8; single block of 128 threads; tc out (128)
__global__ void head_kernel(const float* __restrict__ pooled,
                            const float* __restrict__ pw1, const float* __restrict__ pb1,
                            const float* __restrict__ pw2, const float* __restrict__ pb2,
                            const float* __restrict__ wih, const float* __restrict__ whh,
                            const float* __restrict__ bih, const float* __restrict__ bhh,
                            float* __restrict__ tc) {
    __shared__ float sp[128], shid[128], sseq[T_][128], hbuf[128], cbuf[128];
    int d = threadIdx.x;
    for (int t = 0; t < T_; ++t) {
        sp[d] = pooled[t * 128 + d];
        __syncthreads();
        float a = pb1[d];
        for (int k = 0; k < 128; ++k) a += sp[k] * pw1[k * 128 + d];
        shid[d] = a > 0.0f ? a : 0.0f;
        __syncthreads();
        float b = pb2[d];
        for (int k = 0; k < 128; ++k) b += shid[k] * pw2[k * 128 + d];
        sseq[t][d] = b;
        __syncthreads();
    }
    for (int l = 0; l < 2; ++l) {
        const float* Wih = wih + (size_t)l * 512 * 128;
        const float* Whh = whh + (size_t)l * 512 * 128;
        const float* Bih = bih + l * 512;
        const float* Bhh = bhh + l * 512;
        hbuf[d] = 0.0f; cbuf[d] = 0.0f;
        __syncthreads();
        for (int t = 0; t < T_; ++t) {
            float gi = Bih[d]       + Bhh[d];
            float gf = Bih[128 + d] + Bhh[128 + d];
            float gg = Bih[256 + d] + Bhh[256 + d];
            float go = Bih[384 + d] + Bhh[384 + d];
            for (int k = 0; k < 128; ++k) {
                float xk = sseq[t][k], hk = hbuf[k];
                gi += xk * Wih[(0 * 128 + d) * 128 + k] + hk * Whh[(0 * 128 + d) * 128 + k];
                gf += xk * Wih[(1 * 128 + d) * 128 + k] + hk * Whh[(1 * 128 + d) * 128 + k];
                gg += xk * Wih[(2 * 128 + d) * 128 + k] + hk * Whh[(2 * 128 + d) * 128 + k];
                go += xk * Wih[(3 * 128 + d) * 128 + k] + hk * Whh[(3 * 128 + d) * 128 + k];
            }
            float cn = sigmf(gf) * cbuf[d] + sigmf(gi) * tanhf(gg);
            float hn = sigmf(go) * tanhf(cn);
            __syncthreads();
            hbuf[d] = hn; cbuf[d] = cn; sseq[t][d] = hn;
            __syncthreads();
        }
    }
    tc[d] = sseq[T_ - 1][d];
}

// Build edge feature rows (f16): [src+dst | |src-dst| | tc] width 384; zero pad rows
__global__ void edge_gather_kernel(const int* __restrict__ pairs,
                                   const half_t* __restrict__ ne,
                                   const float* __restrict__ tc,
                                   half_t* __restrict__ ef,
                                   int base, int rowsPad, int P) {
    long idx = (long)blockIdx.x * 256 + threadIdx.x;
    if (idx >= (long)rowsPad * 384) return;
    int r = (int)(idx / 384), c = (int)(idx % 384);
    int e = base + r;
    float val = 0.0f;
    if (e < P) {
        if (c < 256) {
            int c0 = c & 127;
            int i = pairs[2 * e], j = pairs[2 * e + 1];
            float s  = (float)ne[(size_t)i * 128 + c0];
            float dd = (float)ne[(size_t)j * 128 + c0];
            val = (c < 128) ? (s + dd) : fabsf(s - dd);
        } else {
            val = tc[c - 256];
        }
    }
    ef[(size_t)r * 384 + c] = (half_t)val;
}

__global__ void edge_out_kernel(const half_t* __restrict__ z2,
                                const float* __restrict__ w3, const float* __restrict__ b3,
                                float* __restrict__ out, int base, int rows) {
    int r = blockIdx.x * 256 + threadIdx.x;
    if (r >= rows) return;
    const half_t* zr = z2 + (size_t)r * 64;
    float s = b3[0];
#pragma unroll
    for (int k = 0; k < 64; ++k) s += (float)zr[k] * w3[k];
    out[base + r] = s;
}

// ---------------------------------------------------------------------------
// Host-side orchestration
// ---------------------------------------------------------------------------
extern "C" void kernel_launch(void* const* d_in, const int* in_sizes, int n_in,
                              void* d_out, int out_size, void* d_ws, size_t ws_size,
                              hipStream_t stream) {
    const float* node_features = (const float*)d_in[0];
    const float* adj     = (const float*)d_in[1];
    const float* enc_w1  = (const float*)d_in[2];
    const float* enc_b1  = (const float*)d_in[3];
    const float* enc_w2  = (const float*)d_in[4];
    const float* enc_b2  = (const float*)d_in[5];
    const float* msg_w1  = (const float*)d_in[6];
    const float* msg_b1  = (const float*)d_in[7];
    const float* msg_w2  = (const float*)d_in[8];
    const float* msg_b2  = (const float*)d_in[9];
    const float* upd_w1  = (const float*)d_in[10];
    const float* upd_b1  = (const float*)d_in[11];
    const float* upd_w2  = (const float*)d_in[12];
    const float* upd_b2  = (const float*)d_in[13];
    const float* ln_g    = (const float*)d_in[14];
    const float* ln_b    = (const float*)d_in[15];
    const float* pool_w1 = (const float*)d_in[16];
    const float* pool_b1 = (const float*)d_in[17];
    const float* pool_w2 = (const float*)d_in[18];
    const float* pool_b2 = (const float*)d_in[19];
    const float* lstm_wih = (const float*)d_in[20];
    const float* lstm_whh = (const float*)d_in[21];
    const float* lstm_bih = (const float*)d_in[22];
    const float* lstm_bhh = (const float*)d_in[23];
    const float* ep_w1   = (const float*)d_in[24];
    const float* ep_b1   = (const float*)d_in[25];
    const float* ep_w2   = (const float*)d_in[26];
    const float* ep_b2   = (const float*)d_in[27];
    const float* ep_w3   = (const float*)d_in[28];
    const float* ep_b3   = (const float*)d_in[29];
    const int*   edge_pairs = (const int*)d_in[30];
    float* out = (float*)d_out;
    const int P = in_sizes[30] / 2;

    // workspace bump allocator (256B aligned)
    char* wsb = (char*)d_ws;
    size_t off = 0;
    auto alloc = [&](size_t bytes) -> void* {
        void* p = wsb + off;
        off = (off + bytes + 255) & ~(size_t)255;
        return p;
    };

    half_t* pos_h  = (half_t*)alloc((size_t)N_ * PD_ * 2);
    // big region: x(TNx128) / mi(TNx192) / ui(TNx256) / ef(ECx384)
    size_t bigBytes = (size_t)TN_ * 256 * 2;
    size_t efBytes  = (size_t)EC_ * 384 * 2;
    half_t* big_h  = (half_t*)alloc(bigBytes > efBytes ? bigBytes : efBytes);
    half_t* tmp1_h = (half_t*)alloc((size_t)TN_ * 128 * 2);   // also z1 (ECx128)
    float*  h_f    = (float*) alloc((size_t)TN_ * 128 * 4);
    half_t* h_h    = (half_t*)alloc((size_t)TN_ * 128 * 2);
    half_t* m_h    = (half_t*)alloc((size_t)TN_ * 128 * 2);
    half_t* agg_h  = (half_t*)alloc((size_t)TN_ * 128 * 2);   // also z2 (ECx64)
    float*  u_f    = (float*) alloc((size_t)TN_ * 128 * 4);
    half_t* enc_w1h = (half_t*)alloc(128 * 128 * 2);
    half_t* enc_w2h = (half_t*)alloc(128 * 128 * 2);
    half_t* msg_w1h = (half_t*)alloc(2 * 192 * 128 * 2);
    half_t* msg_w2h = (half_t*)alloc(2 * 128 * 128 * 2);
    half_t* upd_w1h = (half_t*)alloc(2 * 256 * 128 * 2);
    half_t* upd_w2h = (half_t*)alloc(2 * 128 * 128 * 2);
    half_t* ep_w1h  = (half_t*)alloc(384 * 128 * 2);
    half_t* ep_w2h  = (half_t*)alloc(128 * 64 * 2);
    float*  pooled  = (float*)alloc(T_ * 128 * 4);
    float*  tc      = (float*)alloc(128 * 4);

    auto cvt = [&](const float* s, half_t* d2, int n) {
        cvt_f16_kernel<<<(n + 255) / 256, 256, 0, stream>>>(s, d2, n);
    };
    cvt(enc_w1, enc_w1h, 128 * 128);
    cvt(enc_w2, enc_w2h, 128 * 128);
    cvt(msg_w1, msg_w1h, 2 * 192 * 128);
    cvt(msg_w2, msg_w2h, 2 * 128 * 128);
    cvt(upd_w1, upd_w1h, 2 * 256 * 128);
    cvt(upd_w2, upd_w2h, 2 * 128 * 128);
    cvt(ep_w1,  ep_w1h, 384 * 128);
    cvt(ep_w2,  ep_w2h, 128 * 64);

    pos_kernel<<<(N_ * PD_ + 255) / 256, 256, 0, stream>>>(pos_h);

    // Encoder
    concat_enc_kernel<<<(TN_ * 128 + 255) / 256, 256, 0, stream>>>(node_features, pos_h, big_h);
    gemm_f16_kernel<8><<<TN_ / 128, 256, 0, stream>>>(big_h, 128, enc_w1h, enc_b1,
                                                      nullptr, tmp1_h, TN_, 128, 1);
    gemm_f16_kernel<8><<<TN_ / 128, 256, 0, stream>>>(tmp1_h, 128, enc_w2h, enc_b2,
                                                      h_f, h_h, TN_, 128, 0);

    // GNN layers
    for (int l = 0; l < 2; ++l) {
        concat_hpos_kernel<<<(int)(((long)TN_ * 192 + 255) / 256), 256, 0, stream>>>(h_h, pos_h, big_h);
        gemm_f16_kernel<8><<<TN_ / 128, 256, 0, stream>>>(big_h, 192, msg_w1h + (size_t)l * 192 * 128,
                                                          msg_b1 + l * 128, nullptr, tmp1_h, TN_, 192, 1);
        gemm_f16_kernel<8><<<TN_ / 128, 256, 0, stream>>>(tmp1_h, 128, msg_w2h + (size_t)l * 128 * 128,
                                                          msg_b2 + l * 128, nullptr, m_h, TN_, 128, 0);
        agg_kernel<<<dim3(N_ / 128, T_), 256, 0, stream>>>(adj, m_h, agg_h);
        concat_hagg_kernel<<<(int)(((long)TN_ * 256 + 255) / 256), 256, 0, stream>>>(h_h, agg_h, big_h);
        gemm_f16_kernel<8><<<TN_ / 128, 256, 0, stream>>>(big_h, 256, upd_w1h + (size_t)l * 256 * 128,
                                                          upd_b1 + l * 128, nullptr, tmp1_h, TN_, 256, 1);
        gemm_f16_kernel<8><<<TN_ / 128, 256, 0, stream>>>(tmp1_h, 128, upd_w2h + (size_t)l * 128 * 128,
                                                          upd_b2 + l * 128, u_f, nullptr, TN_, 128, 0);
        ln_residual_kernel<<<TN_ / 8, 256, 0, stream>>>(u_f, h_f, h_h, ln_g + l * 128, ln_b + l * 128);
    }

    // Pool + MLP + LSTM head
    pool_kernel<<<T_, 128, 0, stream>>>(h_f, pooled);
    head_kernel<<<1, 128, 0, stream>>>(pooled, pool_w1, pool_b1, pool_w2, pool_b2,
                                       lstm_wih, lstm_whh, lstm_bih, lstm_bhh, tc);

    // Edge predictor, chunked
    const half_t* ne = h_h + (size_t)(T_ - 1) * N_ * 128;
    for (int base = 0; base < P; base += EC_) {
        int rows = P - base; if (rows > EC_) rows = EC_;
        int rowsPad = (rows + 127) & ~127;
        edge_gather_kernel<<<(int)(((long)rowsPad * 384 + 255) / 256), 256, 0, stream>>>(
            edge_pairs, ne, tc, big_h, base, rowsPad, P);
        gemm_f16_kernel<8><<<rowsPad / 128, 256, 0, stream>>>(big_h, 384, ep_w1h, ep_b1,
                                                              nullptr, tmp1_h, rows, 384, 1);
        gemm_f16_kernel<4><<<rowsPad / 128, 256, 0, stream>>>(tmp1_h, 128, ep_w2h, ep_b2,
                                                              nullptr, agg_h, rows, 128, 1);
        edge_out_kernel<<<(rows + 255) / 256, 256, 0, stream>>>(agg_h, ep_w3, ep_b3, out, base, rows);
    }
}
